// Block_10453950399093
// MI455X (gfx1250) — matrix operand order repaired and checked by reading, not compiled, not run on hardware
//
#include <hip/hip_runtime.h>
#include <cstdint>

typedef __bf16 bf16_t;
typedef bf16_t v8bf  __attribute__((ext_vector_type(8)));
typedef bf16_t v16bf __attribute__((ext_vector_type(16)));
typedef float  v8f   __attribute__((ext_vector_type(8)));
typedef unsigned int v4u __attribute__((ext_vector_type(4)));
typedef int v8i __attribute__((ext_vector_type(8)));
typedef int v4i __attribute__((ext_vector_type(4)));

constexpr int kB = 4, kT = 2048, kC = 1024, kH = 16, kD = 64, kFF = 4096;
constexpr int kBT = kB * kT;
constexpr int k3C = 3 * kC;

// Padded LDS row stride for staged 32-element bf16 tile rows:
// 64B data + 16B pad = 80B -> keeps b128 alignment, conflict-free fragment reads.
constexpr int kRowStride = 80;

__device__ __forceinline__ v16bf cat8(v8bf lo, v8bf hi) {
  return __builtin_shufflevector(lo, hi, 0, 1, 2, 3, 4, 5, 6, 7, 8, 9, 10, 11, 12, 13, 14, 15);
}

// A-fragment (16x32 bf16, row-major source, leading dim lda in elements):
// lane m holds row m; K groups {0..7,16..23} (lanes 0-15) / {8..15,24..31} (lanes 16-31).
__device__ __forceinline__ v16bf load_a_frag(const bf16_t* a, int lda, int lane) {
  const bf16_t* p = a + (size_t)(lane & 15) * lda + ((lane >> 4) << 3);
  v8bf lo = *(const v8bf*)p;
  v8bf hi = *(const v8bf*)(p + 16);
  return cat8(lo, hi);
}

// B-fragment (32x16 bf16) from K-contiguous (transposed) source: bt[n][k], ldb = K.
__device__ __forceinline__ v16bf load_b_frag(const bf16_t* bt, int ldb, int lane) {
  const bf16_t* p = bt + (size_t)(lane & 15) * ldb + ((lane >> 4) << 3);
  v8bf lo = *(const v8bf*)p;
  v8bf hi = *(const v8bf*)(p + 16);
  return cat8(lo, hi);
}

__device__ __forceinline__ v8f wmma_bf16(v16bf a, v16bf b, v8f c) {
  return __builtin_amdgcn_wmma_f32_16x16x32_bf16(false, a, false, b, (short)0, c, false, false);
}

// ---- Tensor Data Mover: stage a (rows x 32) bf16 tile (row stride = k_len elems)
// from global into LDS with 16B row padding (D# per CDNA5 ISA 8.3/8.4).
__device__ __forceinline__ void tdm_stage_tile(uint64_t gaddr, uint32_t lds_addr,
                                               int k_len, int rows) {
  v4u g0;
  g0.x = 1u;                                  // count=1, user mode
  g0.y = lds_addr;                            // LDS byte address
  g0.z = (uint32_t)gaddr;                     // global_addr[31:0]
  g0.w = (uint32_t)((gaddr >> 32) & 0x01FFFFFFu) | (2u << 30);  // addr[56:32] | type=2
  v8i g1;
  // workgroup_mask=0 | data_size=1(2B)<<16 | pad_enable<<20 | pad_interval=3(16DW)<<22
  // | pad_amount=3(4DW)<<25
  g1[0] = (int)((1u << 16) | (1u << 20) | (3u << 22) | (3u << 25));
  g1[1] = (int)(((uint32_t)k_len & 0xFFFFu) << 16);            // tensor_dim0[15:0]
  g1[2] = (int)((((uint32_t)k_len >> 16) & 0xFFFFu) |          // tensor_dim0[31:16]
                (((uint32_t)rows & 0xFFFFu) << 16));           // tensor_dim1[15:0]
  g1[3] = (int)((((uint32_t)rows >> 16) & 0xFFFFu) |           // tensor_dim1[31:16]
                (32u << 16));                                  // tile_dim0 = 32
  g1[4] = (int)((uint32_t)rows & 0xFFFFu);                     // tile_dim1 = rows, tile_dim2=0
  g1[5] = (int)(uint32_t)k_len;                                // tensor_dim0_stride[31:0]
  g1[6] = 0;                                                   // stride hi | dim1_stride lo
  g1[7] = 0;
  v4i z4;
  z4[0] = 0; z4[1] = 0; z4[2] = 0; z4[3] = 0;
  v8i z8;
#pragma unroll
  for (int i = 0; i < 8; ++i) z8[i] = 0;
  // 6-arg form (clang-23 / therock headers): (g0, g1, g2, g3, extra, cpol)
  __builtin_amdgcn_tensor_load_to_lds(g0, g1, z4, z4, z8, 0);
}

enum { EP_BF16 = 0, EP_BIAS_RES_F32 = 1, EP_BIAS_RELU_BF16 = 2 };

// C = A(bf16, MxK row-major) @ Bt^T (Bt is NxK row-major). Block = 4 waves;
// block tile 64(M) x 256(N); each wave computes 64x64 (16 WMMA accumulators).
// A/B k-tiles staged via double-buffered TDM into padded LDS.
template <int MODE>
__global__ __launch_bounds__(128) void gemm_wmma_kernel(
    const bf16_t* __restrict__ A, const bf16_t* __restrict__ Bt,
    const float* __restrict__ bias, const float* __restrict__ resid,
    void* __restrict__ out, int M, int N, int K) {
  const int lane = threadIdx.x & 31;
  const int wave = threadIdx.x >> 5;
  const int m0 = blockIdx.y * 64;
  const int nblk = blockIdx.x * 256;
  const int nw = nblk + wave * 64;

  __shared__ __align__(16) char ldsA[2][64 * kRowStride];
  __shared__ __align__(16) char ldsB[2][256 * kRowStride];

  v8f acc[16];
#pragma unroll
  for (int t = 0; t < 16; ++t)
#pragma unroll
    for (int i = 0; i < 8; ++i) acc[t][i] = 0.0f;

  const uint64_t gA = (uint64_t)(uintptr_t)(A + (size_t)m0 * K);
  const uint64_t gB = (uint64_t)(uintptr_t)(Bt + (size_t)nblk * K);
  const uint32_t ldsAaddr[2] = {(uint32_t)(uintptr_t)&ldsA[0][0],
                                (uint32_t)(uintptr_t)&ldsA[1][0]};
  const uint32_t ldsBaddr[2] = {(uint32_t)(uintptr_t)&ldsB[0][0],
                                (uint32_t)(uintptr_t)&ldsB[1][0]};

  const int nsteps = K / 32;
  if (wave == 0) {
    tdm_stage_tile(gA, ldsAaddr[0], K, 64);
    tdm_stage_tile(gB, ldsBaddr[0], K, 256);
  }
  for (int i = 0; i < nsteps; ++i) {
    const int cur = i & 1;
    if (i + 1 < nsteps) {
      if (wave == 0) {
        const uint64_t koff = (uint64_t)(i + 1) * 64;  // 32 elems * 2 bytes
        tdm_stage_tile(gA + koff, ldsAaddr[cur ^ 1], K, 64);
        tdm_stage_tile(gB + koff, ldsBaddr[cur ^ 1], K, 256);
        __builtin_amdgcn_s_wait_tensorcnt(2);  // current pair done; next pair in flight
      }
    } else {
      if (wave == 0) __builtin_amdgcn_s_wait_tensorcnt(0);
    }
    __syncthreads();

    const char* pa = &ldsA[cur][0];
    const char* pb = &ldsB[cur][0];
    v16bf af[4];
#pragma unroll
    for (int m = 0; m < 4; ++m) {
      const char* p = pa + (size_t)(m * 16 + (lane & 15)) * kRowStride + ((lane >> 4) << 4);
      af[m] = cat8(*(const v8bf*)p, *(const v8bf*)(p + 32));
    }
#pragma unroll
    for (int n = 0; n < 4; ++n) {
      const char* p = pb + (size_t)(wave * 64 + n * 16 + (lane & 15)) * kRowStride +
                      ((lane >> 4) << 4);
      v16bf bfg = cat8(*(const v8bf*)p, *(const v8bf*)(p + 32));
#pragma unroll
      for (int m = 0; m < 4; ++m) acc[m * 4 + n] = wmma_bf16(af[m], bfg, acc[m * 4 + n]);
    }
    __syncthreads();
  }

  // C/D layout: lane holds column; VGPR r -> row r (+8 for lanes >= 16).
  const int col = lane & 15;
  const int rbase = (lane >> 4) << 3;
#pragma unroll
  for (int m = 0; m < 4; ++m) {
#pragma unroll
    for (int n = 0; n < 4; ++n) {
#pragma unroll
      for (int r = 0; r < 8; ++r) {
        const int row = m0 + m * 16 + rbase + r;
        const int cc = nw + n * 16 + col;
        const size_t idx = (size_t)row * N + cc;
        float v = acc[m * 4 + n][r];
        if constexpr (MODE == EP_BF16) {
          ((bf16_t*)out)[idx] = (bf16_t)v;
        } else if constexpr (MODE == EP_BIAS_RES_F32) {
          ((float*)out)[idx] = v + bias[cc] + resid[idx];
        } else {  // EP_BIAS_RELU_BF16
          v += bias[cc];
          v = v > 0.0f ? v : 0.0f;
          ((bf16_t*)out)[idx] = (bf16_t)v;
        }
      }
    }
  }
}

__global__ __launch_bounds__(256) void layernorm_bf16_kernel(
    const float* __restrict__ x, const float* __restrict__ g,
    const float* __restrict__ b, bf16_t* __restrict__ out) {
  const int row = blockIdx.x;
  const float* xr = x + (size_t)row * kC;
  float s = 0.f, s2 = 0.f;
  for (int i = threadIdx.x; i < kC; i += 256) {
    const float v = xr[i];
    s += v;
    s2 += v * v;
  }
#pragma unroll
  for (int m = 16; m >= 1; m >>= 1) {
    s += __shfl_xor(s, m, 32);
    s2 += __shfl_xor(s2, m, 32);
  }
  __shared__ float sh[2][8];
  if ((threadIdx.x & 31) == 0) {
    sh[0][threadIdx.x >> 5] = s;
    sh[1][threadIdx.x >> 5] = s2;
  }
  __syncthreads();
  float ts = 0.f, ts2 = 0.f;
#pragma unroll
  for (int i = 0; i < 8; ++i) {
    ts += sh[0][i];
    ts2 += sh[1][i];
  }
  const float mu = ts * (1.0f / kC);
  const float var = ts2 * (1.0f / kC) - mu * mu;
  const float rstd = rsqrtf(var + 1e-5f);
  bf16_t* orow = out + (size_t)row * kC;
  for (int i = threadIdx.x; i < kC; i += 256) {
    orow[i] = (bf16_t)((xr[i] - mu) * rstd * g[i] + b[i]);
  }
}

// W (KxN f32 row-major) -> Wt (NxK bf16 row-major)
__global__ void transpose_f32_to_bf16_kernel(const float* __restrict__ W,
                                             bf16_t* __restrict__ Wt, int K, int N) {
  const size_t idx = (size_t)blockIdx.x * 256 + threadIdx.x;
  if (idx >= (size_t)K * N) return;
  const int k = (int)(idx / N);
  const int n = (int)(idx % N);
  Wt[(size_t)n * K + k] = (bf16_t)W[idx];
}

// vt[((b*H+h)*D + d)*T + t] = V[b,t,h,d] taken from qkv (third chunk)
__global__ void build_vt_kernel(const bf16_t* __restrict__ qkv, bf16_t* __restrict__ vt) {
  const size_t idx = (size_t)blockIdx.x * 256 + threadIdx.x;
  if (idx >= (size_t)kB * kH * kD * kT) return;
  const int t = (int)(idx % kT);
  size_t rest = idx / kT;
  const int d = (int)(rest % kD);
  rest /= kD;
  const int h = (int)(rest % kH);
  const int b = (int)(rest / kH);
  vt[idx] = qkv[(size_t)(b * kT + t) * k3C + 2 * kC + h * kD + d];
}

// Flash attention: one wave per (b, h, 16-row Q tile). 32-key chunks, online softmax.
__global__ __launch_bounds__(32) void flash_attn_kernel(
    const bf16_t* __restrict__ qkv, const bf16_t* __restrict__ vt,
    bf16_t* __restrict__ y) {
  const int lane = threadIdx.x;
  const int q0 = blockIdx.x * 16;
  const int h = blockIdx.y;
  const int b = blockIdx.z;
  const int hb = lane >> 4;
  const int col = lane & 15;
  __shared__ bf16_t Pt[16 * 32];  // P tile staged through LDS for C/D -> A layout transpose

  const bf16_t* qrow = qkv + (size_t)(b * kT + q0) * k3C + h * kD;
  const v16bf aq0 = load_a_frag(qrow, k3C, lane);       // d 0..31
  const v16bf aq1 = load_a_frag(qrow + 32, k3C, lane);  // d 32..63

  v8f o[4];
#pragma unroll
  for (int t = 0; t < 4; ++t)
#pragma unroll
    for (int i = 0; i < 8; ++i) o[t][i] = 0.f;
  float mrow[8], lrow[8];
#pragma unroll
  for (int r = 0; r < 8; ++r) {
    mrow[r] = -1e30f;
    lrow[r] = 0.f;
  }

  const bf16_t* kbase = qkv + (size_t)(b * kT) * k3C + kC + h * kD;
  const bf16_t* vtb = vt + (size_t)((b * kH + h) * kD) * kT;  // [d][t]
  const int kend = q0 + 15;

  for (int kc = 0; kc <= kend; kc += 32) {
    v8f s0, s1;
#pragma unroll
    for (int i = 0; i < 8; ++i) {
      s0[i] = 0.f;
      s1[i] = 0.f;
    }
    {
      const bf16_t* kp = kbase + (size_t)kc * k3C;
      s0 = wmma_bf16(aq0, load_b_frag(kp, k3C, lane), s0);
      s0 = wmma_bf16(aq1, load_b_frag(kp + 32, k3C, lane), s0);
      kp += (size_t)16 * k3C;
      s1 = wmma_bf16(aq0, load_b_frag(kp, k3C, lane), s1);
      s1 = wmma_bf16(aq1, load_b_frag(kp + 32, k3C, lane), s1);
    }
#pragma unroll
    for (int r = 0; r < 8; ++r) {
      const int qi = q0 + (hb << 3) + r;  // this lane-half's query row
      float v0 = s0[r] * 0.125f;          // 1/sqrt(64)
      float v1 = s1[r] * 0.125f;
      if (kc + col > qi) v0 = -1e30f;       // causal mask, key tile 0
      if (kc + 16 + col > qi) v1 = -1e30f;  // causal mask, key tile 1
      float cmax = fmaxf(v0, v1);
#pragma unroll
      for (int m = 8; m >= 1; m >>= 1) cmax = fmaxf(cmax, __shfl_xor(cmax, m, 32));
      const float mnew = fmaxf(mrow[r], cmax);
      const float corr = __expf(mrow[r] - mnew);
      const float p0 = __expf(v0 - mnew);
      const float p1 = __expf(v1 - mnew);
      float rs = p0 + p1;
#pragma unroll
      for (int m = 8; m >= 1; m >>= 1) rs += __shfl_xor(rs, m, 32);
      lrow[r] = lrow[r] * corr + rs;
      mrow[r] = mnew;
#pragma unroll
      for (int t = 0; t < 4; ++t) o[t][r] *= corr;
      const int prow = (hb << 3) + r;
      Pt[prow * 32 + col] = (bf16_t)p0;
      Pt[prow * 32 + 16 + col] = (bf16_t)p1;
    }
    __syncthreads();
    const v16bf ap = load_a_frag(Pt, 32, lane);  // P as A-fragment (16x32)
    __syncthreads();
#pragma unroll
    for (int t = 0; t < 4; ++t) {
      const bf16_t* vp = vtb + (size_t)(t * 16) * kT + kc;  // columns d = t*16.., rows = keys
      o[t] = wmma_bf16(ap, load_b_frag(vp, kT, lane), o[t]);
    }
  }

  bf16_t* yrow = y + (size_t)(b * kT + q0) * kC + h * kD;
#pragma unroll
  for (int r = 0; r < 8; ++r) {
    const float inv = 1.0f / lrow[r];
#pragma unroll
    for (int t = 0; t < 4; ++t) {
      yrow[(size_t)((hb << 3) + r) * kC + t * 16 + col] = (bf16_t)(o[t][r] * inv);
    }
  }
}

extern "C" void kernel_launch(void* const* d_in, const int* in_sizes, int n_in,
                              void* d_out, int out_size, void* d_ws, size_t ws_size,
                              hipStream_t stream) {
  (void)in_sizes; (void)n_in; (void)out_size; (void)ws_size;
  const float* x      = (const float*)d_in[0];
  const float* W_attn = (const float*)d_in[1];
  const float* W_proj = (const float*)d_in[2];
  const float* b_proj = (const float*)d_in[3];
  const float* W_fc1  = (const float*)d_in[4];
  const float* b_fc1  = (const float*)d_in[5];
  const float* W_fc2  = (const float*)d_in[6];
  const float* b_fc2  = (const float*)d_in[7];
  const float* ln1_g  = (const float*)d_in[8];
  const float* ln1_b  = (const float*)d_in[9];
  const float* ln2_g  = (const float*)d_in[10];
  const float* ln2_b  = (const float*)d_in[11];

  char* ws = (char*)d_ws;
  size_t off = 0;
  auto alloc = [&](size_t bytes) -> void* {
    void* p = ws + off;
    off = (off + bytes + 255) & ~(size_t)255;
    return p;
  };
  bf16_t* WattnT = (bf16_t*)alloc((size_t)k3C * kC * 2);
  bf16_t* WprojT = (bf16_t*)alloc((size_t)kC * kC * 2);
  bf16_t* Wfc1T  = (bf16_t*)alloc((size_t)kFF * kC * 2);
  bf16_t* Wfc2T  = (bf16_t*)alloc((size_t)kC * kFF * 2);
  bf16_t* hbuf   = (bf16_t*)alloc((size_t)kBT * kC * 2);
  bf16_t* qkvb   = (bf16_t*)alloc((size_t)kBT * k3C * 2);
  bf16_t* vtb    = (bf16_t*)alloc((size_t)kB * kH * kD * kT * 2);
  bf16_t* ybuf   = (bf16_t*)alloc((size_t)kBT * kC * 2);
  float*  x1     = (float*)alloc((size_t)kBT * kC * 4);
  bf16_t* h2buf  = (bf16_t*)alloc((size_t)kBT * kC * 2);
  bf16_t* m1buf  = (bf16_t*)alloc((size_t)kBT * kFF * 2);

  // 1. weight convert+transpose to bf16 [N][K]
  { size_t n = (size_t)kC * k3C; transpose_f32_to_bf16_kernel<<<(n + 255) / 256, 256, 0, stream>>>(W_attn, WattnT, kC, k3C); }
  { size_t n = (size_t)kC * kC;  transpose_f32_to_bf16_kernel<<<(n + 255) / 256, 256, 0, stream>>>(W_proj, WprojT, kC, kC); }
  { size_t n = (size_t)kC * kFF; transpose_f32_to_bf16_kernel<<<(n + 255) / 256, 256, 0, stream>>>(W_fc1, Wfc1T, kC, kFF); }
  { size_t n = (size_t)kFF * kC; transpose_f32_to_bf16_kernel<<<(n + 255) / 256, 256, 0, stream>>>(W_fc2, Wfc2T, kFF, kC); }

  // 2. h = LN1(x) -> bf16
  layernorm_bf16_kernel<<<kBT, 256, 0, stream>>>(x, ln1_g, ln1_b, hbuf);

  // 3. qkv = h @ W_attn (bf16 out)
  gemm_wmma_kernel<EP_BF16><<<dim3(k3C / 256, kBT / 64), 128, 0, stream>>>(
      hbuf, WattnT, nullptr, nullptr, qkvb, kBT, k3C, kC);

  // 4. V transposed per head: [b,h,d,t]
  { size_t n = (size_t)kB * kH * kD * kT; build_vt_kernel<<<(n + 255) / 256, 256, 0, stream>>>(qkvb, vtb); }

  // 5. causal flash attention -> y (bf16)
  flash_attn_kernel<<<dim3(kT / 16, kH, kB), 32, 0, stream>>>(qkvb, vtb, ybuf);

  // 6. x1 = x + y @ W_proj + b_proj (f32)
  gemm_wmma_kernel<EP_BIAS_RES_F32><<<dim3(kC / 256, kBT / 64), 128, 0, stream>>>(
      ybuf, WprojT, b_proj, x, x1, kBT, kC, kC);

  // 7. h2 = LN2(x1) -> bf16
  layernorm_bf16_kernel<<<kBT, 256, 0, stream>>>(x1, ln2_g, ln2_b, h2buf);

  // 8. m1 = relu(h2 @ W_fc1 + b_fc1) -> bf16
  gemm_wmma_kernel<EP_BIAS_RELU_BF16><<<dim3(kFF / 256, kBT / 64), 128, 0, stream>>>(
      h2buf, Wfc1T, b_fc1, nullptr, m1buf, kBT, kFF, kC);

  // 9. out = x1 + m1 @ W_fc2 + b_fc2 (f32, final output)
  gemm_wmma_kernel<EP_BIAS_RES_F32><<<dim3(kC / 256, kBT / 64), 128, 0, stream>>>(
      m1buf, Wfc2T, b_fc2, x1, (float*)d_out, kBT, kC, kFF);
}
